// MultiwayMLP_4810363372427
// MI455X (gfx1250) — compile-verified, matrix operand order checked by reading, not run on hardware
//
#include <hip/hip_runtime.h>
#include <hip/hip_bf16.h>
#include <math.h>

// Problem constants (from reference)
#define BB 2
#define SS 2048
#define DD 768
#define EE 4
#define LL 2
#define HH (4*DD)           // 3072
#define NTOK (BB*SS)        // 4096
#define TILE_M 64
#define NTILES_M 68         // 68*64 = 4352 >= 4096 + 4*63 (segment padding)
#define NTOK_PAD (NTILES_M*TILE_M)
#define EPSF 1e-6f

// Workspace layout (bytes)
#define WS_PERM_OFF   64ull
#define WS_H_OFF      17664ull                       // 4352*768  bf16 rows
#define WS_A_OFF      6702336ull                     // 4352*3072 bf16 rows
#define WS_W1_OFF     33441024ull                    // L*E*D*H bf16 (tiled)
#define WS_W2_OFF     71189760ull                    // L*E*H*D bf16 (tiled)
#define WS_NEED_CACHE 108938496ull

typedef __attribute__((ext_vector_type(16))) __bf16 v16bf;
typedef __attribute__((ext_vector_type(8)))  float  v8f;

__device__ __forceinline__ unsigned short f2bf(float f) {
  unsigned int u = __builtin_bit_cast(unsigned int, f);
  u += 0x7FFFu + ((u >> 16) & 1u);          // round-to-nearest-even
  return (unsigned short)(u >> 16);
}

// ------- one-time: convert W f32 -> bf16, transposed+tiled for WMMA B-operand -------
// Output layout per (l,e) chunk of K*N elems: tiles [nb][kb][col 0..127][k 0..31],
// i.e. exactly the LDS image the GEMM wants -> GEMM B staging is a straight copy.
template<int K, int N>
__global__ __launch_bounds__(256)
void k_cvtW(const float* __restrict__ src, unsigned short* __restrict__ dst) {
  __shared__ __align__(16) unsigned short lds[128 * 32];
  int kb = blockIdx.x, nb = blockIdx.y, le = blockIdx.z;
  int tid = threadIdx.x;
  int r   = tid >> 3;             // 0..31 (k within tile)
  int c16 = (tid & 7) << 4;       // 16 cols per thread
  const float* s = src + (size_t)le * K * N + (size_t)(kb * 32 + r) * N + nb * 128 + c16;
  #pragma unroll
  for (int i4 = 0; i4 < 4; ++i4) {
    float4 f = ((const float4*)s)[i4];
    lds[(c16 + i4 * 4 + 0) * 32 + r] = f2bf(f.x);
    lds[(c16 + i4 * 4 + 1) * 32 + r] = f2bf(f.y);
    lds[(c16 + i4 * 4 + 2) * 32 + r] = f2bf(f.z);
    lds[(c16 + i4 * 4 + 3) * 32 + r] = f2bf(f.w);
  }
  __syncthreads();
  unsigned short* d = dst + ((size_t)le * K * N)
                    + ((size_t)nb * (K / 32) + kb) * 4096 + tid * 16;
  ((uint4*)d)[0] = *(const uint4*)&lds[tid * 16];
  ((uint4*)d)[1] = *(const uint4*)&lds[tid * 16 + 8];
}

// ---------------- routing ----------------
__global__ void k_init(int* cnt, int* perm) {
  int i = blockIdx.x * blockDim.x + threadIdx.x;
  if (i < 8) cnt[i] = 0;
  if (i < NTOK_PAD) perm[i] = -1;
}

__global__ void k_count(const int* __restrict__ eidx, int* cnt) {
  int t = blockIdx.x * blockDim.x + threadIdx.x;
  if (t < NTOK) atomicAdd(&cnt[eidx[t]], 1);
}

__global__ void k_offsets(int* cnt, int* off) {
  if (threadIdx.x == 0 && blockIdx.x == 0) {
    int acc = 0;
    for (int e = 0; e < EE; ++e) {
      off[e] = acc;
      acc += (cnt[e] + TILE_M - 1) & ~(TILE_M - 1);   // 64-align segments
      cnt[e] = 0;                                      // reuse as cursor
    }
    off[EE] = acc;
  }
}

__global__ void k_scatter(const int* __restrict__ eidx, int* cnt,
                          const int* __restrict__ off, int* perm) {
  int t = blockIdx.x * blockDim.x + threadIdx.x;
  if (t < NTOK) {
    int e = eidx[t];
    int p = off[e] + atomicAdd(&cnt[e], 1);
    perm[p] = t;
  }
}

// ---------------- layernorm -> compacted bf16 rows ----------------
__global__ __launch_bounds__(256)
void k_layernorm(const float* __restrict__ xcur, const int* __restrict__ perm,
                 const float* __restrict__ g, const float* __restrict__ bta,
                 unsigned short* __restrict__ h_c) {
  int row = blockIdx.x;
  int tid = threadIdx.x;
  int tok = perm[row];
  unsigned short* hr = h_c + (size_t)row * DD;
  if (tok < 0) {                       // pad row: deterministic zeros
    #pragma unroll
    for (int j = 0; j < 3; ++j) hr[tid + j * 256] = 0;
    return;
  }
  const float* xr = xcur + (size_t)tok * DD;
  float v[3], s = 0.f, sq = 0.f;
  #pragma unroll
  for (int j = 0; j < 3; ++j) {
    v[j] = xr[tid + j * 256];
    s += v[j]; sq += v[j] * v[j];
  }
  __shared__ float red[256];
  red[tid] = s; __syncthreads();
  for (int w = 128; w > 0; w >>= 1) { if (tid < w) red[tid] += red[tid + w]; __syncthreads(); }
  float mu = red[0] * (1.0f / DD);
  __syncthreads();
  red[tid] = sq; __syncthreads();
  for (int w = 128; w > 0; w >>= 1) { if (tid < w) red[tid] += red[tid + w]; __syncthreads(); }
  float var  = red[0] * (1.0f / DD) - mu * mu;
  float rstd = rsqrtf(var + EPSF);
  #pragma unroll
  for (int j = 0; j < 3; ++j) {
    int i = tid + j * 256;
    hr[i] = f2bf((v[j] - mu) * rstd * g[i] + bta[i]);
  }
}

// ---------------- grouped GEMM (bf16 WMMA, f32 accumulate) ----------------
// Block: 256 threads = 8 waves. Tile: 64 M x 128 N, K-step 32.
// Each wave: 1 B fragment (16 cols) reused by 4 A fragments -> 4 WMMAs / k-step.
// BF16W path: B tiles are pre-transposed in memory -> staging is a straight copy.
template<int K, int N, bool GELU, bool BF16W>
__global__ __launch_bounds__(256)
void k_moe_gemm(const unsigned short* __restrict__ Abuf, // [NTOK_PAD][K] bf16
                const float* __restrict__ Wf,            // [E][K][N] f32 (fallback)
                const unsigned short* __restrict__ Wh,   // [E][tiled] bf16 (cached)
                const float* __restrict__ biasl,         // [E][N] f32
                const int* __restrict__ off,
                const int* __restrict__ perm,
                unsigned short* __restrict__ gout,       // GELU: [NTOK_PAD][N] bf16
                float* __restrict__ xout)                // !GELU: [NTOK][N] f32 (+=)
{
  __shared__ __align__(16) unsigned short lds_a[TILE_M * 32];  // 4 KB
  __shared__ __align__(16) unsigned short lds_b[128 * 32];     // 8 KB, [col][k]

  int m0 = blockIdx.x * TILE_M;
  if (m0 >= off[EE]) return;                 // block-uniform
  int e = 0;
  #pragma unroll
  for (int q = 1; q < EE; ++q) e += (m0 >= off[q]);

  const float*          Wf_e = Wf ? Wf + (size_t)e * K * N : nullptr;
  const unsigned short* Wh_e = Wh ? Wh + (size_t)e * K * N : nullptr;
  const float* bias = biasl + (size_t)e * N;
  int n0   = blockIdx.y * 128;
  int tid  = threadIdx.x;
  int wave = tid >> 5, lane = tid & 31;

  v8f c[4] = {{}, {}, {}, {}};
  int sra = tid >> 2;             // 0..63 (A staging row)
  int sca = (tid & 3) << 3;       // 8 bf16 per thread
  int srb = tid >> 3;             // 0..31 (B staging k-row, fallback)
  int scb = (tid & 7) << 4;       // 16 cols per thread (fallback)
  // cached path: consecutive K-step tiles are contiguous (8 KB apart)
  const unsigned short* btile =
      BF16W ? (Wh_e + ((size_t)(n0 >> 7) * (K / 32)) * 4096 + tid * 16) : nullptr;

  for (int k0 = 0; k0 < K; k0 += 32) {
    // stage A tile (64 x 32 bf16), coalesced 16B per thread
    *(uint4*)&lds_a[sra * 32 + sca] =
        *(const uint4*)(Abuf + (size_t)(m0 + sra) * K + k0 + sca);
    // stage B tile (already [col][k] in memory on the cached path)
    if constexpr (BF16W) {
      *(uint4*)&lds_b[tid * 16]     = ((const uint4*)btile)[0];
      *(uint4*)&lds_b[tid * 16 + 8] = ((const uint4*)btile)[1];
      btile += 4096;
    } else {
      const float* src = Wf_e + (size_t)(k0 + srb) * N + n0 + scb;
      #pragma unroll
      for (int i4 = 0; i4 < 4; ++i4) {
        float4 f = ((const float4*)src)[i4];
        lds_b[(scb + i4 * 4 + 0) * 32 + srb] = f2bf(f.x);
        lds_b[(scb + i4 * 4 + 1) * 32 + srb] = f2bf(f.y);
        lds_b[(scb + i4 * 4 + 2) * 32 + srb] = f2bf(f.z);
        lds_b[(scb + i4 * 4 + 3) * 32 + srb] = f2bf(f.w);
      }
    }
    __syncthreads();

    // Fragments per ISA 16-bit layouts: lane<16 -> K {0..7,16..23}; lane>=16 -> {8..15,24..31}
    int khalf = (lane >> 4) << 3;
    int am    = lane & 15;
    union FR { v16bf v; uint4 q[2]; } fa[4], fb;
    int bc = wave * 16 + am;
    fb.q[0] = *(const uint4*)&lds_b[bc * 32 + khalf];
    fb.q[1] = *(const uint4*)&lds_b[bc * 32 + khalf + 16];
    #pragma unroll
    for (int s = 0; s < 4; ++s) {
      fa[s].q[0] = *(const uint4*)&lds_a[(s * 16 + am) * 32 + khalf];
      fa[s].q[1] = *(const uint4*)&lds_a[(s * 16 + am) * 32 + khalf + 16];
    }
    #pragma unroll
    for (int s = 0; s < 4; ++s)
      c[s] = __builtin_amdgcn_wmma_f32_16x16x32_bf16(false, fa[s].v, false, fb.v,
                                                     (short)0, c[s], false, false);
    __syncthreads();
  }

  // Epilogue: C/D layout -> (m = 8*(lane>>4)+r, n = lane&15)
  int nl = lane & 15;
  int mh = (lane >> 4) << 3;
  int n  = n0 + wave * 16 + nl;
  float bv = bias[n];
  #pragma unroll
  for (int s = 0; s < 4; ++s) {
    #pragma unroll
    for (int r = 0; r < 8; ++r) {
      int row = m0 + s * 16 + mh + r;
      float v = c[s][r] + bv;
      if (GELU) {
        float gv = 0.5f * v * (1.0f + erff(v * 0.70710678118654752f));
        gout[(size_t)row * N + n] = f2bf(gv);
      } else {
        int tok = perm[row];
        if (tok >= 0) xout[(size_t)tok * N + n] += v;   // unique (tok,n) per lane
      }
    }
  }
}

// ---------------- host orchestration ----------------
extern "C" void kernel_launch(void* const* d_in, const int* in_sizes, int n_in,
                              void* d_out, int out_size, void* d_ws, size_t ws_size,
                              hipStream_t stream) {
  const float* x    = (const float*)d_in[0];
  const int*   eidx = (const int*)  d_in[1];
  const float* W1   = (const float*)d_in[2];
  const float* b1   = (const float*)d_in[3];
  const float* W2   = (const float*)d_in[4];
  const float* b2   = (const float*)d_in[5];
  const float* ln_g = (const float*)d_in[6];
  const float* ln_b = (const float*)d_in[7];
  float* xcur = (float*)d_out;              // running activation == final output

  char* ws = (char*)d_ws;
  int* cnt  = (int*)ws;
  int* off  = (int*)(ws + 32);
  int* perm = (int*)(ws + WS_PERM_OFF);
  unsigned short* h_c  = (unsigned short*)(ws + WS_H_OFF);
  unsigned short* a_bf = (unsigned short*)(ws + WS_A_OFF);
  unsigned short* w1bf = (unsigned short*)(ws + WS_W1_OFF);
  unsigned short* w2bf = (unsigned short*)(ws + WS_W2_OFF);

  const bool cacheW = ws_size >= WS_NEED_CACHE;     // fixed per process -> deterministic

  hipMemcpyAsync(xcur, x, sizeof(float) * (size_t)NTOK * DD,
                 hipMemcpyDeviceToDevice, stream);

  if (cacheW) {
    k_cvtW<DD, HH><<<dim3(DD / 32, HH / 128, LL * EE), 256, 0, stream>>>(W1, w1bf);
    k_cvtW<HH, DD><<<dim3(HH / 32, DD / 128, LL * EE), 256, 0, stream>>>(W2, w2bf);
  }

  k_init   <<<(NTOK_PAD + 255) / 256, 256, 0, stream>>>(cnt, perm);
  k_count  <<<NTOK / 256, 256, 0, stream>>>(eidx, cnt);
  k_offsets<<<1, 1, 0, stream>>>(cnt, off);
  k_scatter<<<NTOK / 256, 256, 0, stream>>>(eidx, cnt, off, perm);

  for (int l = 0; l < LL; ++l) {
    k_layernorm<<<NTOK_PAD, 256, 0, stream>>>(
        xcur, perm, ln_g + (size_t)l * DD, ln_b + (size_t)l * DD, h_c);

    if (cacheW) {
      k_moe_gemm<DD, HH, true, true><<<dim3(NTILES_M, HH / 128), 256, 0, stream>>>(
          h_c, nullptr, w1bf + (size_t)l * EE * DD * HH,
          b1 + (size_t)l * EE * HH, off, perm, a_bf, nullptr);
      k_moe_gemm<HH, DD, false, true><<<dim3(NTILES_M, DD / 128), 256, 0, stream>>>(
          a_bf, nullptr, w2bf + (size_t)l * EE * HH * DD,
          b2 + (size_t)l * EE * DD, off, perm, nullptr, xcur);
    } else {
      k_moe_gemm<DD, HH, true, false><<<dim3(NTILES_M, HH / 128), 256, 0, stream>>>(
          h_c, W1 + (size_t)l * EE * DD * HH, nullptr,
          b1 + (size_t)l * EE * HH, off, perm, a_bf, nullptr);
      k_moe_gemm<HH, DD, false, false><<<dim3(NTILES_M, DD / 128), 256, 0, stream>>>(
          a_bf, W2 + (size_t)l * EE * HH * DD, nullptr,
          b2 + (size_t)l * EE * DD, off, perm, nullptr, xcur);
    }
  }
}
